// RNN_model_50543175140032
// MI455X (gfx1250) — matrix operand order repaired and checked by reading, not compiled, not run on hardware
//
#include <hip/hip_runtime.h>

// ---------------------------------------------------------------------------
// Fused 2-layer tanh RNN (B=256, T=2048, OBS=64, H=128) + 3-layer MLP head.
// 16 workgroups x 256 threads (8 waves, wave32). Each workgroup owns 16 batch
// rows for the entire sequence; h-state double-buffered in LDS; all weight
// WMMA B-fragments hoisted to registers; v_wmma_f32_16x16x32_bf16 with f32
// accumulation; native v_tanh_f32; x loads software-pipelined one step ahead.
// ---------------------------------------------------------------------------

typedef __attribute__((ext_vector_type(16))) __bf16       v16bf;
typedef __attribute__((ext_vector_type(8)))  float        v8f;
typedef __attribute__((ext_vector_type(4)))  unsigned int v4u;
typedef __attribute__((ext_vector_type(4)))  float        v4f;

#define OBS_D 64
#define ACT_D 32
#define H_D   128
#define B_D   256
#define T_D   2048
#define XP    72    // x tile row pitch (bf16) — padded for LDS banks
#define HP    136   // H-sized row pitch (bf16) — padded for LDS banks

// LDS element offsets (in __bf16 units); every segment 16B-aligned.
#define OFF_WIH0 0                      // 128*72   = 9216
#define OFF_WHH0 9216                   // 128*136  = 17408
#define OFF_WIH1 26624
#define OFF_WHH1 44032
#define OFF_X    61440                  // 16*72    = 1152
#define OFF_H0   62592                  // 2*16*136 = 4352
#define OFF_H1   66944
#define SMEM_ELEMS 71296
#define SMEM_BYTES (SMEM_ELEMS * 2)     // 142592 bytes (< 320KB/WGP)

// Native gfx1250 v_tanh_f32 if the toolchain exposes it; libm fallback keeps
// the compile green either way.
#if __has_builtin(__builtin_amdgcn_tanhf)
#define FAST_TANH(x) __builtin_amdgcn_tanhf(x)
#elif __has_builtin(__builtin_amdgcn_tanh_f32)
#define FAST_TANH(x) __builtin_amdgcn_tanh_f32(x)
#else
#define FAST_TANH(x) tanhf(x)
#endif

__device__ __forceinline__ v8f wmma_bf16(v16bf a, v16bf b, v8f c) {
  // emits v_wmma_f32_16x16x32_bf16
  return __builtin_amdgcn_wmma_f32_16x16x32_bf16(false, a, false, b, (short)0, c,
                                                 false, false);
}

// A-matrix 16x32 bf16 fragment: lane m=lane&15, h=lane>>4:
// row m, K=[16h,16h+16) contiguous (matches B fragment order).
__device__ __forceinline__ v16bf load_a_frag(const __bf16* base, int pitch,
                                             int k0, int lane) {
  const int m = lane & 15, hh = lane >> 4;
  const __bf16* p = base + m * pitch + k0;
  union { v16bf v; v4u q[2]; } u;
  u.q[0] = *(const v4u*)(p + 16 * hh);
  u.q[1] = *(const v4u*)(p + 16 * hh + 8);
  return u.v;
}

// B-matrix 32x16 bf16 fragment: lane n=lane&15, h=lane>>4 holds column n,
// K=[16h,16h+16). Since B = W^T, lane n reads W[nrow0+n, k0+16h ..] contiguously.
__device__ __forceinline__ v16bf load_b_frag(const __bf16* wbase, int pitch,
                                             int nrow0, int k0, int lane) {
  const int n = lane & 15, hh = lane >> 4;
  const __bf16* p = wbase + (nrow0 + n) * pitch + k0 + 16 * hh;
  union { v16bf v; v4u q[2]; } u;
  u.q[0] = *(const v4u*)(p);
  u.q[1] = *(const v4u*)(p + 8);
  return u.v;
}

// B fragment converted on the fly from row-major f32 weights in global memory.
__device__ __forceinline__ v16bf load_b_frag_gf32(const float* W, int ld,
                                                  int nrow0, int k0, int lane) {
  const int n = lane & 15, hh = lane >> 4;
  const float* p = W + (size_t)(nrow0 + n) * ld + k0 + 16 * hh;
  v16bf r;
#pragma unroll
  for (int i = 0; i < 16; ++i) r[i] = (__bf16)p[i];
  return r;
}

__global__ void __launch_bounds__(256) rnn2_fused_kernel(
    const float* __restrict__ x,
    const float* __restrict__ Wih0, const float* __restrict__ Whh0,
    const float* __restrict__ bih0, const float* __restrict__ bhh0,
    const float* __restrict__ Wih1, const float* __restrict__ Whh1,
    const float* __restrict__ bih1, const float* __restrict__ bhh1,
    const float* __restrict__ Wf0,  const float* __restrict__ bf0,
    const float* __restrict__ Wf1,  const float* __restrict__ bf1,
    const float* __restrict__ Wf2,  const float* __restrict__ bf2,
    float* __restrict__ out)
{
  extern __shared__ __bf16 smem[];
  __bf16* lWih0 = smem + OFF_WIH0;
  __bf16* lWhh0 = smem + OFF_WHH0;
  __bf16* lWih1 = smem + OFF_WIH1;
  __bf16* lWhh1 = smem + OFF_WHH1;
  __bf16* lX    = smem + OFF_X;
  __bf16* lH0   = smem + OFF_H0;
  __bf16* lH1   = smem + OFF_H1;

  const int tid  = threadIdx.x;
  const int lane = tid & 31;       // wave32
  const int wv   = tid >> 5;       // wave id 0..7 -> owns N-tile [16wv,16wv+16)
  const int b0   = blockIdx.x * 16;
  const int hh   = lane >> 4;
  const int ng   = wv * 16 + (lane & 15);   // global output column 0..127

  // ---- preload weights (f32 -> bf16) into padded LDS, zero h state ----
  for (int i = tid; i < H_D * OBS_D; i += 256)
    lWih0[(i >> 6) * XP + (i & 63)] = (__bf16)Wih0[i];
  for (int i = tid; i < H_D * H_D; i += 256) {
    const int r = i >> 7, c = i & 127;
    lWhh0[r * HP + c] = (__bf16)Whh0[i];
    lWih1[r * HP + c] = (__bf16)Wih1[i];
    lWhh1[r * HP + c] = (__bf16)Whh1[i];
  }
  for (int i = tid; i < 2 * 16 * HP; i += 256) {
    lH0[i] = (__bf16)0.0f;
    lH1[i] = (__bf16)0.0f;
  }
  __syncthreads();

  // ---- hoist all loop-invariant weight B-fragments into registers ----
  v16bf Bih0[2], Bhh0[4], Bih1[4], Bhh1[4];
#pragma unroll
  for (int k = 0; k < 2; ++k) Bih0[k] = load_b_frag(lWih0, XP, wv * 16, k * 32, lane);
#pragma unroll
  for (int k = 0; k < 4; ++k) {
    Bhh0[k] = load_b_frag(lWhh0, HP, wv * 16, k * 32, lane);
    Bih1[k] = load_b_frag(lWih1, HP, wv * 16, k * 32, lane);
    Bhh1[k] = load_b_frag(lWhh1, HP, wv * 16, k * 32, lane);
  }
  const float bias0 = bih0[ng] + bhh0[ng];
  const float bias1 = bih1[ng] + bhh1[ng];

  // x staging: 16 threads per batch row, one float4 each; software-pipelined
  const int xm = tid >> 4;
  const int xc = (tid & 15) * 4;
  const float* xrow = x + (size_t)(b0 + xm) * T_D * OBS_D + xc;
  __bf16* xd = lX + xm * XP + xc;

  v4f xv = *(const v4f*)(xrow);   // prime the pipeline with x_0

  int cur = 0;
  for (int t = 0; t < T_D; ++t) {
    // stage x_t (held in registers), then immediately kick off x_{t+1} load so
    // it has the whole step of WMMA work to cover its latency.
    xd[0] = (__bf16)xv.x; xd[1] = (__bf16)xv.y;
    xd[2] = (__bf16)xv.z; xd[3] = (__bf16)xv.w;
    if (t + 1 < T_D) {
      xv = *(const v4f*)(xrow + (size_t)(t + 1) * OBS_D);
      __builtin_prefetch(xrow + (size_t)(t + 2) * OBS_D, 0, 3);  // WGP-scope
    }
    __syncthreads();

    __bf16* h0c = lH0 + cur * (16 * HP);
    __bf16* h0n = lH0 + (cur ^ 1) * (16 * HP);
    __bf16* h1c = lH1 + cur * (16 * HP);
    __bf16* h1n = lH1 + (cur ^ 1) * (16 * HP);

    // layer 0: h0' = tanh(x_t @ Wih0^T + b0 + h0 @ Whh0^T)
    // issue all A-fragment loads up front, then chain the WMMAs
    v16bf aX0 = load_a_frag(lX,  XP,  0, lane);
    v16bf aX1 = load_a_frag(lX,  XP, 32, lane);
    v16bf aH0 = load_a_frag(h0c, HP,  0, lane);
    v16bf aH1 = load_a_frag(h0c, HP, 32, lane);
    v16bf aH2 = load_a_frag(h0c, HP, 64, lane);
    v16bf aH3 = load_a_frag(h0c, HP, 96, lane);
    v8f acc = {};
    acc = wmma_bf16(aX0, Bih0[0], acc);
    acc = wmma_bf16(aX1, Bih0[1], acc);
    acc = wmma_bf16(aH0, Bhh0[0], acc);
    acc = wmma_bf16(aH1, Bhh0[1], acc);
    acc = wmma_bf16(aH2, Bhh0[2], acc);
    acc = wmma_bf16(aH3, Bhh0[3], acc);
#pragma unroll
    for (int r = 0; r < 8; ++r)
      h0n[(r + 8 * hh) * HP + ng] = (__bf16)FAST_TANH(acc[r] + bias0);
    __syncthreads();

    // layer 1: h1' = tanh(h0' @ Wih1^T + b1 + h1 @ Whh1^T)
    v16bf aI0 = load_a_frag(h0n, HP,  0, lane);
    v16bf aI1 = load_a_frag(h0n, HP, 32, lane);
    v16bf aI2 = load_a_frag(h0n, HP, 64, lane);
    v16bf aI3 = load_a_frag(h0n, HP, 96, lane);
    v16bf aR0 = load_a_frag(h1c, HP,  0, lane);
    v16bf aR1 = load_a_frag(h1c, HP, 32, lane);
    v16bf aR2 = load_a_frag(h1c, HP, 64, lane);
    v16bf aR3 = load_a_frag(h1c, HP, 96, lane);
    v8f acc1 = {};
    acc1 = wmma_bf16(aI0, Bih1[0], acc1);
    acc1 = wmma_bf16(aI1, Bih1[1], acc1);
    acc1 = wmma_bf16(aI2, Bih1[2], acc1);
    acc1 = wmma_bf16(aI3, Bih1[3], acc1);
    acc1 = wmma_bf16(aR0, Bhh1[0], acc1);
    acc1 = wmma_bf16(aR1, Bhh1[1], acc1);
    acc1 = wmma_bf16(aR2, Bhh1[2], acc1);
    acc1 = wmma_bf16(aR3, Bhh1[3], acc1);
#pragma unroll
    for (int r = 0; r < 8; ++r)
      h1n[(r + 8 * hh) * HP + ng] = (__bf16)FAST_TANH(acc1[r] + bias1);

    cur ^= 1;
  }
  __syncthreads();

  const __bf16* h0f = lH0 + cur * (16 * HP);
  const __bf16* h1f = lH1 + cur * (16 * HP);

  // hn output: d_out[8192 + layer*32768 + b*128 + c]
  for (int i = tid; i < 16 * 128; i += 256) {
    const int m = i >> 7, c = i & 127;
    out[8192 + (size_t)(b0 + m) * 128 + c]         = (float)h0f[m * HP + c];
    out[8192 + 32768 + (size_t)(b0 + m) * 128 + c] = (float)h1f[m * HP + c];
  }

  // ---- MLP head on h1_T (reuses free h buffers as scratch) ----
  __bf16* m1 = lH0 + (cur ^ 1) * (16 * HP);
  __bf16* m2 = lH1 + (cur ^ 1) * (16 * HP);

  v8f a0 = {};
#pragma unroll
  for (int k = 0; k < 4; ++k)
    a0 = wmma_bf16(load_a_frag(h1f, HP, k * 32, lane),
                   load_b_frag_gf32(Wf0, 128, wv * 16, k * 32, lane), a0);
  {
    const float bb = bf0[ng];
#pragma unroll
    for (int r = 0; r < 8; ++r)
      m1[(r + 8 * hh) * HP + ng] = (__bf16)FAST_TANH(a0[r] + bb);
  }
  __syncthreads();

  v8f a1 = {};
#pragma unroll
  for (int k = 0; k < 4; ++k)
    a1 = wmma_bf16(load_a_frag(m1, HP, k * 32, lane),
                   load_b_frag_gf32(Wf1, 128, wv * 16, k * 32, lane), a1);
  {
    const float bb = bf1[ng];
#pragma unroll
    for (int r = 0; r < 8; ++r)
      m2[(r + 8 * hh) * HP + ng] = (__bf16)FAST_TANH(a1[r] + bb);
  }
  __syncthreads();

  if (wv < 2) {  // wave-uniform branch: EXEC stays all-ones for WMMA
    v8f a2 = {};
#pragma unroll
    for (int k = 0; k < 4; ++k)
      a2 = wmma_bf16(load_a_frag(m2, HP, k * 32, lane),
                     load_b_frag_gf32(Wf2, 128, wv * 16, k * 32, lane), a2);
    const float bb = bf2[ng];
#pragma unroll
    for (int r = 0; r < 8; ++r)
      out[(size_t)(b0 + r + 8 * hh) * ACT_D + ng] = a2[r] + bb;
  }
}

extern "C" void kernel_launch(void* const* d_in, const int* in_sizes, int n_in,
                              void* d_out, int out_size, void* d_ws, size_t ws_size,
                              hipStream_t stream) {
  (void)in_sizes; (void)n_in; (void)out_size; (void)d_ws; (void)ws_size;
  const float* x    = (const float*)d_in[0];
  const float* Wih0 = (const float*)d_in[1];
  const float* Whh0 = (const float*)d_in[2];
  const float* bih0 = (const float*)d_in[3];
  const float* bhh0 = (const float*)d_in[4];
  const float* Wih1 = (const float*)d_in[5];
  const float* Whh1 = (const float*)d_in[6];
  const float* bih1 = (const float*)d_in[7];
  const float* bhh1 = (const float*)d_in[8];
  const float* Wf0  = (const float*)d_in[9];
  const float* bf0  = (const float*)d_in[10];
  const float* Wf1  = (const float*)d_in[11];
  const float* bf1  = (const float*)d_in[12];
  const float* Wf2  = (const float*)d_in[13];
  const float* bf2  = (const float*)d_in[14];
  float* out = (float*)d_out;

  rnn2_fused_kernel<<<dim3(B_D / 16), dim3(256), SMEM_BYTES, stream>>>(
      x, Wih0, Whh0, bih0, bhh0, Wih1, Whh1, bih1, bhh1,
      Wf0, bf0, Wf1, bf1, Wf2, bf2, out);
}
